// Head_3874060501298
// MI455X (gfx1250) — compile-verified, matrix operand order checked by reading, not run on hardware
//
#include <hip/hip_runtime.h>

#define N_EMBD 1024
#define HSZ    64
#define TSEQ   4096
#define WPAD   40   // padded per-column k-stride in halves (80B) -> conflict-free b128 LDS reads

typedef __attribute__((ext_vector_type(16))) _Float16 v16h;
typedef __attribute__((ext_vector_type(8)))  _Float16 v8h;
typedef __attribute__((ext_vector_type(8)))  float    v8f;
typedef __attribute__((ext_vector_type(4)))  float    v4f;
typedef __attribute__((ext_vector_type(4)))  int      v4i;

typedef __attribute__((address_space(1))) v4i* g4_t;        // __device__ int4*
typedef __attribute__((address_space(3))) v4i* l4_t;        // __shared__ int4*
typedef __attribute__((address_space(3))) _Float16* lash_t;

#if __has_builtin(__builtin_amdgcn_global_load_async_to_lds_b128)
  #define HAVE_ASYNC_LDS 1
#else
  #define HAVE_ASYNC_LDS 0
#endif

static __device__ __forceinline__ void wait_async0() {
#if __has_builtin(__builtin_amdgcn_s_wait_asynccnt)
  __builtin_amdgcn_s_wait_asynccnt(0);
#else
  asm volatile("s_wait_asynccnt 0" ::: "memory");
#endif
}

static __device__ __forceinline__ v8f wmma16x16x32(v16h a, v16h b, v8f c) {
  return __builtin_amdgcn_wmma_f32_16x16x32_f16(false, a, false, b, (short)0, c, false, false);
}

static __device__ __forceinline__ v16h cat8(v8h lo, v8h hi) {
  return __builtin_shufflevector(lo, hi, 0,1,2,3,4,5,6,7,8,9,10,11,12,13,14,15);
}

static __device__ __forceinline__ v16h loadA(const float* __restrict__ xrow, int kc, int off) {
  v4f f0 = *(const v4f*)(xrow + kc + off);
  v4f f1 = *(const v4f*)(xrow + kc + off + 4);
  v4f f2 = *(const v4f*)(xrow + kc + 16 + off);
  v4f f3 = *(const v4f*)(xrow + kc + 16 + off + 4);
  v16h a;
  #pragma unroll
  for (int i = 0; i < 4; ++i) {
    a[i]      = (_Float16)f0[i];
    a[4 + i]  = (_Float16)f1[i];
    a[8 + i]  = (_Float16)f2[i];
    a[12 + i] = (_Float16)f3[i];
  }
  return a;
}

// ---------------------------------------------------------------------------
// Kernel 0: weights fp32 [1024 x 64] -> fp16 transposed [64 x 1024]
// ---------------------------------------------------------------------------
__global__ void __launch_bounds__(256)
transpose_w(const float* __restrict__ Wk, const float* __restrict__ Wq,
            const float* __restrict__ Wv,
            _Float16* __restrict__ WtK, _Float16* __restrict__ WtQ,
            _Float16* __restrict__ WtV)
{
  int idx  = blockIdx.x * blockDim.x + threadIdx.x;   // 3 * 65536 threads
  int wsel = idx >> 16;
  int rem  = idx & 65535;
  int h    = rem & 63;
  int kc   = rem >> 6;
  const float* W = (wsel == 0) ? Wk : (wsel == 1) ? Wq : Wv;
  _Float16*   Wt = (wsel == 0) ? WtK : (wsel == 1) ? WtQ : WtV;
  Wt[h * N_EMBD + kc] = (_Float16)W[kc * HSZ + h];
}

// ---------------------------------------------------------------------------
// Kernel 1: q/k/v projections.  4 waves/WG, 16 rows each.  Weight K-chunks
// (12 KB) staged in double-buffered LDS via async global->LDS copies shared
// by all 4 waves; x A-operand double-buffered in registers; all 12 B
// fragments materialized before the WMMA burst so DS loads pipeline.
// ---------------------------------------------------------------------------
__global__ void __launch_bounds__(128)
proj_qkv(const float* __restrict__ x,
         const _Float16* __restrict__ WtK, const _Float16* __restrict__ WtQ,
         const _Float16* __restrict__ WtV,
         _Float16* __restrict__ qh, _Float16* __restrict__ kh,
         _Float16* __restrict__ vT)
{
  __shared__ __align__(16) _Float16 Wlds[2][3 * 64 * WPAD];   // 30 KB

  const int tid  = threadIdx.x;
  const int lane = tid & 31;
  const int wv   = tid >> 5;
  const int half = lane >> 4;
  const int ln   = lane & 15;
  const int off  = half * 8;
  const int rows = (blockIdx.x * 4 + wv) * 16;     // base row in [0, B*T)

  v8f accq[4] = {}, acck[4] = {}, accv[4] = {};
  const float* xrow = x + (size_t)(rows + ln) * N_EMBD;

  // This thread's 6 16-byte staging transfers per K-chunk.
  // j = u*128 + tid ; m = j>>8 (0:Q 1:K 2:V) ; c = (j&255)>>2 ; seg = j&3
  const _Float16* wsrc[6];
  int dsth[6];
  #pragma unroll
  for (int u = 0; u < 6; ++u) {
    int j   = u * 128 + tid;
    int mm  = j >> 8;
    int c   = (j & 255) >> 2;
    int seg = j & 3;
    const _Float16* Wm = (mm == 0) ? WtQ : (mm == 1) ? WtK : WtV;
    wsrc[u] = Wm + (size_t)c * N_EMBD + seg * 8;
    dsth[u] = (mm * 64 + c) * WPAD + seg * 8;
  }

#if HAVE_ASYNC_LDS
  auto stage = [&](int pn, int kc) {
    lash_t lb = (lash_t)&Wlds[pn][0];
    #pragma unroll
    for (int u = 0; u < 6; ++u)
      __builtin_amdgcn_global_load_async_to_lds_b128(
          (g4_t)(wsrc[u] + kc), (l4_t)(lb + dsth[u]), 0, 0);
  };
  stage(0, 0);
  wait_async0();
  __syncthreads();
#else
  v8h wtmp[6];
  auto stage_ld = [&](int kc) {
    #pragma unroll
    for (int u = 0; u < 6; ++u) wtmp[u] = *(const v8h*)(wsrc[u] + kc);
  };
  auto stage_st = [&](int pn) {
    #pragma unroll
    for (int u = 0; u < 6; ++u) *(v8h*)&Wlds[pn][dsth[u]] = wtmp[u];
  };
  stage_ld(0); stage_st(0);
  __syncthreads();
#endif

  v16h a = loadA(xrow, 0, off);

  for (int kc = 0; kc < N_EMBD; kc += 32) {
    const int  p    = (kc >> 5) & 1;
    const bool more = (kc + 32) < N_EMBD;

#if HAVE_ASYNC_LDS
    if (more) stage(p ^ 1, kc + 32);     // async copies overlap the WMMAs below
#else
    if (more) stage_ld(kc + 32);         // global loads overlap the WMMAs below
#endif
    v16h an = a;
    if (more) an = loadA(xrow, kc + 32, off);

    // Materialize all 12 B fragments first: 24 ds_load_b128 issue as a
    // pipelined burst, then the 12 WMMAs drain them with partial waits.
    v16h bfr[12];
    #pragma unroll
    for (int i = 0; i < 4; ++i) {
      const _Float16* bq = &Wlds[p][(0 * 64 + i * 16 + ln) * WPAD];
      const _Float16* bk = &Wlds[p][(1 * 64 + i * 16 + ln) * WPAD];
      const _Float16* bv = &Wlds[p][(2 * 64 + i * 16 + ln) * WPAD];
      bfr[i]     = cat8(*(const v8h*)(bq + off), *(const v8h*)(bq + 16 + off));
      bfr[4 + i] = cat8(*(const v8h*)(bk + off), *(const v8h*)(bk + 16 + off));
      bfr[8 + i] = cat8(*(const v8h*)(bv + off), *(const v8h*)(bv + 16 + off));
    }
    #pragma unroll
    for (int i = 0; i < 4; ++i) {
      accq[i] = wmma16x16x32(a, bfr[i],     accq[i]);
      acck[i] = wmma16x16x32(a, bfr[4 + i], acck[i]);
      accv[i] = wmma16x16x32(a, bfr[8 + i], accv[i]);
    }

#if HAVE_ASYNC_LDS
    if (more) wait_async0();
#else
    if (more) stage_st(p ^ 1);
#endif
    __syncthreads();
    a = an;
  }

  const int b  = rows / TSEQ;
  const int tl = rows - b * TSEQ;
  #pragma unroll
  for (int i = 0; i < 4; ++i) {
    #pragma unroll
    for (int r = 0; r < 8; ++r) {
      size_t row = (size_t)(rows + r + 8 * half);
      qh[row * HSZ + i * 16 + ln] = (_Float16)accq[i][r];
      kh[row * HSZ + i * 16 + ln] = (_Float16)acck[i][r];
    }
    v8h pv;
    #pragma unroll
    for (int r = 0; r < 8; ++r) pv[r] = (_Float16)accv[i][r];
    *(v8h*)(vT + ((size_t)b * HSZ + i * 16 + ln) * TSEQ + tl + 8 * half) = pv;
  }
}

// ---------------------------------------------------------------------------
// Kernel 2: causal flash attention.  One wave per 16-query tile; 32-key
// blocks; next block's K/V fragments register double-buffered.
// ---------------------------------------------------------------------------
__global__ void __launch_bounds__(128)
attn_flash(const _Float16* __restrict__ qh, const _Float16* __restrict__ kh,
           const _Float16* __restrict__ vT, float* __restrict__ out)
{
  __shared__ __align__(16) _Float16 Plds[4][16 * WPAD];

  const int lane  = threadIdx.x & 31;
  const int wv    = threadIdx.x >> 5;
  const int half  = lane >> 4;
  const int ln    = lane & 15;
  const int off   = half * 8;
  const int wid   = blockIdx.x * 4 + wv;             // 0 .. B*T/16-1
  const int b     = wid >> 8;                        // 256 tiles per batch
  const int qbase = (wid & 255) * 16;
  const int nk    = qbase + 16;

  const _Float16* qrow = qh + ((size_t)b * TSEQ + qbase + ln) * HSZ;
  v16h qa0 = cat8(*(const v8h*)(qrow + off),      *(const v8h*)(qrow + 16 + off));
  v16h qa1 = cat8(*(const v8h*)(qrow + 32 + off), *(const v8h*)(qrow + 48 + off));

  const _Float16* khb = kh + (size_t)b * TSEQ * HSZ + (size_t)ln * HSZ;
  const _Float16* vb  = vT + (size_t)b * HSZ * TSEQ + (size_t)ln * TSEQ;

  auto load_kv = [&](int kb, v16h kk[4], v16h vv[4]) {
    const _Float16* kr0 = khb + (size_t)kb * HSZ;
    const _Float16* kr1 = kr0 + 16 * HSZ;
    kk[0] = cat8(*(const v8h*)(kr0 + off),      *(const v8h*)(kr0 + 16 + off));
    kk[1] = cat8(*(const v8h*)(kr0 + 32 + off), *(const v8h*)(kr0 + 48 + off));
    kk[2] = cat8(*(const v8h*)(kr1 + off),      *(const v8h*)(kr1 + 16 + off));
    kk[3] = cat8(*(const v8h*)(kr1 + 32 + off), *(const v8h*)(kr1 + 48 + off));
    #pragma unroll
    for (int i = 0; i < 4; ++i) {
      const _Float16* vp = vb + (size_t)i * 16 * TSEQ + kb + off;
      vv[i] = cat8(*(const v8h*)vp, *(const v8h*)(vp + 16));
    }
  };

  v8f o0 = {}, o1 = {}, o2 = {}, o3 = {};
  float m[8], l[8];
  #pragma unroll
  for (int r = 0; r < 8; ++r) { m[r] = -1e30f; l[r] = 0.f; }
  const float scale = 0.03125f;                      // 1024^-0.5

  _Float16* Pw = &Plds[wv][0];

  v16h kcur[4], vcur[4], knxt[4], vnxt[4];
  load_kv(0, kcur, vcur);

  for (int kb = 0; kb < nk; kb += 32) {
    const bool more = (kb + 32) < nk;
    if (more) load_kv(kb + 32, knxt, vnxt);          // overlaps everything below

    // ---- S = Q K^T for two 16-key tiles -------------------------------
    v8f s0 = {}, s1 = {};
    s0 = wmma16x16x32(qa0, kcur[0], s0);
    s0 = wmma16x16x32(qa1, kcur[1], s0);
    s1 = wmma16x16x32(qa0, kcur[2], s1);
    s1 = wmma16x16x32(qa1, kcur[3], s1);

    #pragma unroll
    for (int r = 0; r < 8; ++r) { s0[r] *= scale; s1[r] *= scale; }
    if (kb + 31 > qbase) {                           // causal mask near diagonal
      #pragma unroll
      for (int r = 0; r < 8; ++r) {
        int row = qbase + r + 8 * half;
        if (kb + ln      > row) s0[r] = -1e30f;
        if (kb + 16 + ln > row) s1[r] = -1e30f;
      }
    }

    // ---- online softmax (row = r + 8*half, cols across 16-lane halves)
    #pragma unroll
    for (int r = 0; r < 8; ++r) {
      float tmax = fmaxf(s0[r], s1[r]);
      #pragma unroll
      for (int d = 1; d < 16; d <<= 1) tmax = fmaxf(tmax, __shfl_xor(tmax, d, 32));
      float mn  = fmaxf(m[r], tmax);
      float fac = __expf(m[r] - mn);
      float p0  = __expf(s0[r] - mn);
      float p1  = __expf(s1[r] - mn);
      float rs  = p0 + p1;
      #pragma unroll
      for (int d = 1; d < 16; d <<= 1) rs += __shfl_xor(rs, d, 32);
      l[r] = l[r] * fac + rs;
      m[r] = mn;
      o0[r] *= fac; o1[r] *= fac; o2[r] *= fac; o3[r] *= fac;
      Pw[(r + 8 * half) * WPAD + ln]      = (_Float16)p0;
      Pw[(r + 8 * half) * WPAD + 16 + ln] = (_Float16)p1;
    }
    asm volatile("s_wait_dscnt 0" ::: "memory");

    // ---- reload P in the 16x32 A layout, accumulate O += P * V --------
    const _Float16* pr = Pw + ln * WPAD + off;
    v16h pa = cat8(*(const v8h*)pr, *(const v8h*)(pr + 16));

    o0 = wmma16x16x32(pa, vcur[0], o0);
    o1 = wmma16x16x32(pa, vcur[1], o1);
    o2 = wmma16x16x32(pa, vcur[2], o2);
    o3 = wmma16x16x32(pa, vcur[3], o3);

    if (more) {
      #pragma unroll
      for (int i = 0; i < 4; ++i) { kcur[i] = knxt[i]; vcur[i] = vnxt[i]; }
    }
  }

  // ---- epilogue: out = O / l, fp32 row-major ------------------------------
  #pragma unroll
  for (int r = 0; r < 8; ++r) {
    float inv = 1.0f / l[r];
    size_t row = (size_t)b * TSEQ + qbase + r + 8 * half;
    out[row * HSZ + 0 * 16 + ln] = o0[r] * inv;
    out[row * HSZ + 1 * 16 + ln] = o1[r] * inv;
    out[row * HSZ + 2 * 16 + ln] = o2[r] * inv;
    out[row * HSZ + 3 * 16 + ln] = o3[r] * inv;
  }
}

// ---------------------------------------------------------------------------
extern "C" void kernel_launch(void* const* d_in, const int* in_sizes, int n_in,
                              void* d_out, int out_size, void* d_ws, size_t ws_size,
                              hipStream_t stream)
{
  (void)in_sizes; (void)n_in; (void)out_size; (void)ws_size;
  const float* x  = (const float*)d_in[0];
  const float* Wk = (const float*)d_in[1];
  const float* Wq = (const float*)d_in[2];
  const float* Wv = (const float*)d_in[3];
  float* out = (float*)d_out;

  char* ws = (char*)d_ws;
  _Float16* qh  = (_Float16*)(ws);                       // 2 MB
  _Float16* kh  = (_Float16*)(ws + (size_t)2097152);     // 2 MB
  _Float16* vT  = (_Float16*)(ws + (size_t)4194304);     // 2 MB
  _Float16* WtK = (_Float16*)(ws + (size_t)6291456);     // 128 KB each
  _Float16* WtQ = (_Float16*)(ws + (size_t)6291456 + 131072);
  _Float16* WtV = (_Float16*)(ws + (size_t)6291456 + 262144);

  transpose_w<<<768, 256, 0, stream>>>(Wk, Wq, Wv, WtK, WtQ, WtV);
  proj_qkv   <<<256, 128, 0, stream>>>(x, WtK, WtQ, WtV, qh, kh, vT);
  attn_flash <<<256, 128, 0, stream>>>(qh, kh, vT, out);
}
